// LatticeLSTM_87591563035073
// MI455X (gfx1250) — matrix operand order repaired and checked by reading, not compiled, not run on hardware
//
#include <hip/hip_runtime.h>
#include <hip/hip_bf16.h>

// ---------------- WMMA types & helpers (CDNA5 wave32, 16x16x32 bf16) ----------------
typedef __attribute__((ext_vector_type(16))) __bf16 v16bf;
typedef __attribute__((ext_vector_type(8)))  float  v8f;

__device__ __forceinline__ v8f wmma_bf16(v16bf a, v16bf b, v8f c) {
  // (neg_a, A, neg_b, B, c_mod, C, reuse_a, reuse_b)
  return __builtin_amdgcn_wmma_f32_16x16x32_bf16(false, a, false, b, (short)0, c, false, false);
}

// A fragment: 16x32 (MxK). lane<16 holds K=kk*32..+15 ;
// lane>=16 holds K=kk*32+16..+31. rowptr = &A[Mrow][0] (row-major, Kpad stride).
__device__ __forceinline__ v16bf load_a_frag(const __bf16* rowptr, int kk, int lane) {
  const int kb = kk * 32 + ((lane >> 4) << 4);
  v16bf a;
#pragma unroll
  for (int i = 0; i < 8; ++i) { a[i] = rowptr[kb + i]; a[8 + i] = rowptr[kb + 8 + i]; }
  return a;
}

// B fragment: 32x16 (KxN), weights pre-transposed to Bt[N][Kpad] so each lane reads
// 16 contiguous bf16 (32B, aligned). lane<16 -> K=kk*32..+15 ; lane>=16 -> +16.
__device__ __forceinline__ v16bf load_b_frag(const __bf16* rowptr, int kk, int lane) {
  const int kb = kk * 32 + ((lane >> 4) << 4);
  return *(const v16bf*)(rowptr + kb);
}

__device__ __forceinline__ float sigm(float x) { return 1.0f / (1.0f + __expf(-x)); }

// ---------------- Problem sizes ----------------
// B=32 S=512 K=4 DC=DW=100(pad 128) H=256 L=20
// M_bs = B*S = 16384 ; M_w = B*S*K = 65536

// ---------------- Workspace layout (all offsets 256B aligned) ----------------
constexpr size_t SZ_CHZ   = (size_t)16384 * 128 * 2;   // char emb, bf16, padded K=128
constexpr size_t SZ_WEZ   = (size_t)65536 * 128 * 2;   // word emb, bf16, padded K=128
constexpr size_t SZ_ACOEF = (size_t)65536 * 4;         // alpha*mask per word row
constexpr size_t SZ_HAS   = (size_t)16384 * 4;         // any-word-matched per (b,s)
constexpr size_t SZ_WOUTT = (size_t)32 * 512 * 2;      // W_out^T padded to [32][512] bf16

constexpr size_t OFF_CHZ   = 0;
constexpr size_t OFF_WEZ   = OFF_CHZ + SZ_CHZ;
constexpr size_t OFF_ACOEF = OFF_WEZ + SZ_WEZ;
constexpr size_t OFF_HAS   = OFF_ACOEF + SZ_ACOEF;
constexpr size_t OFF_WOUTT = OFF_HAS + SZ_HAS;
constexpr size_t OFF_DIR   = OFF_WOUTT + SZ_WOUTT;

// per-direction region
constexpr size_t SZ_WXHT  = (size_t)1024 * 384 * 2;    // [Wx;Wh]^T -> [1024][384] bf16
constexpr size_t SZ_WIT   = (size_t)256 * 128 * 2;
constexpr size_t SZ_WCWT  = (size_t)256 * 128 * 2;
constexpr size_t SZ_WGT   = (size_t)256 * 256 * 2;
constexpr size_t SZ_WGSUM = (size_t)16384 * 256 * 2;   // bf16 (feeds wgate matmul)
constexpr size_t SZ_WCSUM = (size_t)16384 * 256 * 4;   // f32
constexpr size_t SZ_WGATE = (size_t)16384 * 256 * 4;   // f32
constexpr size_t SZ_HOUT  = (size_t)16384 * 256 * 2;   // bf16 hidden out

constexpr size_t D_WXHT  = 0;
constexpr size_t D_WIT   = D_WXHT + SZ_WXHT;
constexpr size_t D_WCWT  = D_WIT + SZ_WIT;
constexpr size_t D_WGT   = D_WCWT + SZ_WCWT;
constexpr size_t D_WGSUM = D_WGT + SZ_WGT;
constexpr size_t D_WCSUM = D_WGSUM + SZ_WGSUM;
constexpr size_t D_WGATE = D_WCSUM + SZ_WCSUM;
constexpr size_t D_HOUT  = D_WGATE + SZ_WGATE;
constexpr size_t DIR_STRIDE = D_HOUT + SZ_HOUT;

// ---------------- K1: gather embeddings (fp32 -> bf16, pad 100->128) + lattice coefs ----
__global__ __launch_bounds__(128) void prep_kernel(
    const int* __restrict__ char_ids, const int* __restrict__ word_ids,
    const int* __restrict__ word_lens,
    const float* __restrict__ char_emb, const float* __restrict__ word_emb,
    __bf16* __restrict__ chz, __bf16* __restrict__ wez,
    float* __restrict__ acoef, float* __restrict__ hasb)
{
  const int bs = blockIdx.x;     // 0..16383
  const int j  = threadIdx.x;    // 0..127
  const int cid = char_ids[bs];
  chz[(size_t)bs * 128 + j] = (__bf16)((j < 100) ? char_emb[(size_t)cid * 100 + j] : 0.0f);
  float hassum = 0.f;
#pragma unroll
  for (int k = 0; k < 4; ++k) {
    const int widx = bs * 4 + k;
    const int wid = word_ids[widx];
    wez[(size_t)widx * 128 + j] = (__bf16)((j < 100) ? word_emb[(size_t)wid * 100 + j] : 0.0f);
    if (j == 0) {
      const float m = (wid > 0) ? 1.0f : 0.0f;
      acoef[widx] = m * (1.0f / (1.0f + (float)word_lens[widx]));
      hassum += m;
    }
  }
  if (j == 0) hasb[bs] = hassum;
}

// ---------------- K2: pack [Wx;Wh]^T for gates: dst[1024][384] bf16 ----------------
__global__ __launch_bounds__(256) void packWxh_kernel(const float* __restrict__ Wc,
                                                      __bf16* __restrict__ dst)
{
  const int idx = blockIdx.x * 256 + threadIdx.x;   // 1024*384
  if (idx >= 1024 * 384) return;
  const int n = idx / 384, k = idx % 384;
  float v = 0.f;
  if (k < 128) { if (k < 100) v = Wc[(size_t)k * 1024 + n]; }
  else         { v = Wc[(size_t)(100 + (k - 128)) * 1024 + n]; }
  dst[idx] = (__bf16)v;
}

// generic transpose+pad: src[Ksrc][Nsrc] f32 row-major -> dst[Npad][Kpad] bf16
__global__ __launch_bounds__(256) void packT_kernel(const float* __restrict__ src,
                                                    __bf16* __restrict__ dst,
                                                    int Npad, int Nsrc, int Kpad, int Ksrc)
{
  const int idx = blockIdx.x * 256 + threadIdx.x;
  if (idx >= Npad * Kpad) return;
  const int n = idx / Kpad, k = idx % Kpad;
  dst[idx] = (__bf16)((n < Nsrc && k < Ksrc) ? src[(size_t)k * Nsrc + n] : 0.0f);
}

// ---------------- K3: word projections + lattice reduction over K=4 words -------------
// M = 65536 word rows; per block one 16-row M-tile (= 4 (b,s) groups); 16 waves = 16 N-tiles.
__global__ __launch_bounds__(512) void wordproj_kernel(
    const __bf16* __restrict__ wez, const float* __restrict__ acoef,
    const __bf16* __restrict__ WiT, const __bf16* __restrict__ WcwT,
    const float* __restrict__ bi, const float* __restrict__ bcw,
    __bf16* __restrict__ wgsum, float* __restrict__ wcsum)
{
  const int lane = threadIdx.x & 31;
  const int ntile = threadIdx.x >> 5;       // 0..15
  const int mbase = blockIdx.x << 4;        // word-row base
  const __bf16* arow  = wez  + (size_t)(mbase + (lane & 15)) * 128;
  const __bf16* birow = WiT  + (size_t)(ntile * 16 + (lane & 15)) * 128;
  const __bf16* bcrow = WcwT + (size_t)(ntile * 16 + (lane & 15)) * 128;
  v8f acc_i = {}; v8f acc_c = {};
#pragma unroll
  for (int kk = 0; kk < 4; ++kk) {
    v16bf a = load_a_frag(arow, kk, lane);
    acc_i = wmma_bf16(a, load_b_frag(birow, kk, lane), acc_i);
    acc_c = wmma_bf16(a, load_b_frag(bcrow, kk, lane), acc_c);
  }
  const int ncol = ntile * 16 + (lane & 15);
  const float bias_i = bi[ncol], bias_c = bcw[ncol];
  const int mh = (lane >> 4) << 3;          // 0 or 8: first M row held by this lane
  float swg0 = 0.f, swc0 = 0.f, swg1 = 0.f, swc1 = 0.f;
#pragma unroll
  for (int j = 0; j < 8; ++j) {
    const float am = acoef[mbase + mh + j];                 // alpha*mask for this word row
    const float iw = sigm(acc_i[j] + bias_i);
    const float cw = tanhf(acc_c[j] + bias_c);
    const float t = am * iw;
    if (j < 4) { swg0 += t; swc0 += t * cw; } else { swg1 += t; swc1 += t * cw; }
  }
  const int g0 = (mbase + mh) >> 2;         // (b,s) index of first 4-row group
  wgsum[(size_t)g0 * 256 + ncol]       = (__bf16)swg0;
  wcsum[(size_t)g0 * 256 + ncol]       = swc0;
  wgsum[(size_t)(g0 + 1) * 256 + ncol] = (__bf16)swg1;
  wcsum[(size_t)(g0 + 1) * 256 + ncol] = swc1;
}

// ---------------- K4: wgate = sigmoid(wg_sum @ Wg + bg), M=16384 K=256 N=256 ----------
__global__ __launch_bounds__(512) void wgate_kernel(
    const __bf16* __restrict__ wgsum, const __bf16* __restrict__ WgT,
    const float* __restrict__ bg, float* __restrict__ wgate)
{
  const int lane = threadIdx.x & 31;
  const int ntile = threadIdx.x >> 5;
  const int mbase = blockIdx.x << 4;
  const __bf16* arow = wgsum + (size_t)(mbase + (lane & 15)) * 256;
  const __bf16* brow = WgT   + (size_t)(ntile * 16 + (lane & 15)) * 256;
  v8f acc = {};
#pragma unroll
  for (int kk = 0; kk < 8; ++kk)
    acc = wmma_bf16(load_a_frag(arow, kk, lane), load_b_frag(brow, kk, lane), acc);
  const int ncol = ntile * 16 + (lane & 15);
  const float bias = bg[ncol];
  const int mh = (lane >> 4) << 3;
#pragma unroll
  for (int j = 0; j < 8; ++j)
    wgate[(size_t)(mbase + mh + j) * 256 + ncol] = sigm(acc[j] + bias);
}

// ---------------- K5: serial lattice-LSTM recurrence, one workgroup per direction -----
// gates[32,1024] = [x_t(128) | h(256)] @ WxhT ; i,f,o,g are the wave's own 4 accumulators.
__global__ __launch_bounds__(1024) void recurrence_kernel(
    char* __restrict__ ws, const float* __restrict__ bc0, const float* __restrict__ bc1,
    const int* __restrict__ lengths)
{
  const int dir = blockIdx.x;
  char* db = ws + OFF_DIR + (size_t)dir * DIR_STRIDE;
  const __bf16* WxhT  = (const __bf16*)(db + D_WXHT);
  const float*  wgate = (const float*)(db + D_WGATE);
  const float*  wcsum = (const float*)(db + D_WCSUM);
  __bf16*       hout  = (__bf16*)(db + D_HOUT);
  const __bf16* chz   = (const __bf16*)(ws + OFF_CHZ);
  const float*  hasb  = (const float*)(ws + OFF_HAS);
  const float*  bc    = dir ? bc1 : bc0;

  __shared__ __bf16 sh_h[32 * 256];          // shared hidden state (bf16, matmul operand)

  const int tid  = threadIdx.x;
  const int lane = tid & 31;
  const int wave = tid >> 5;                 // 0..31
  const int mtile = wave & 1;                // batch rows 0-15 / 16-31
  const int ncg   = wave >> 1;               // 0..15 column group
  const int ncol  = ncg * 16 + (lane & 15);  // gate-local column 0..255
  const int arow  = mtile * 16 + (lane & 15);          // batch row for A fragment
  const int mh    = mtile * 16 + ((lane >> 4) << 3);   // first C row owned by this lane

  for (int i = tid; i < 32 * 256; i += 1024) sh_h[i] = (__bf16)0.f;
  float creg[8];
#pragma unroll
  for (int j = 0; j < 8; ++j) creg[j] = 0.f;
  int lenj[8];
#pragma unroll
  for (int j = 0; j < 8; ++j) lenj[j] = lengths[mh + j];
  float bcn[4];
#pragma unroll
  for (int q = 0; q < 4; ++q) bcn[q] = bc[q * 256 + ncol];

  const __bf16* hrow  = sh_h + arow * 256;
  const __bf16* brow0 = WxhT + (size_t)ncol * 384;
  __syncthreads();

  for (int t = 0; t < 512; ++t) {
    const int tt = dir ? (511 - t) : t;
    const __bf16* xrow = chz + ((size_t)arow * 512 + tt) * 128;
    v8f acc[4] = {};
#pragma unroll
    for (int q = 0; q < 4; ++q) {            // q = gate (i,f,o,g) -> columns +256*q
      const __bf16* brow = brow0 + (size_t)q * 256 * 384;
#pragma unroll
      for (int kk = 0; kk < 12; ++kk) {      // K = 128 (x) + 256 (h)
        v16bf a = (kk < 4) ? load_a_frag(xrow, kk, lane)
                           : load_a_frag(hrow, kk - 4, lane);
        acc[q] = wmma_bf16(a, load_b_frag(brow, kk, lane), acc[q]);
      }
    }
    __syncthreads();                         // all sh_h reads complete
#pragma unroll
    for (int j = 0; j < 8; ++j) {
      const int b = mh + j;
      const size_t rowi = (size_t)b * 512 + tt;
      const float ic = sigm(acc[0][j] + bcn[0]);
      const float fc = sigm(acc[1][j] + bcn[1]);
      const float oc = sigm(acc[2][j] + bcn[2]);
      const float gc = tanhf(acc[3][j] + bcn[3]);
      float ct = fc * creg[j] + ic * gc;
      if (hasb[rowi] > 0.f) {
        const float wg = wgate[rowi * 256 + ncol];
        ct = (1.f - wg) * ct + wg * wcsum[rowi * 256 + ncol];
      }
      const float ht = oc * tanhf(ct);
      float hn;
      if (tt < lenj[j]) { creg[j] = ct; hn = ht; }
      else              { hn = (float)sh_h[b * 256 + ncol]; }   // hold state past length
      sh_h[b * 256 + ncol] = (__bf16)hn;
      hout[rowi * 256 + ncol] = (__bf16)hn;
    }
    __syncthreads();                         // sh_h updated before next step's matmul
  }
}

// ---------------- K6: emissions = [h_f | h_b] @ W_out + b_out  (M=16384 K=512 N=20->32)
__global__ __launch_bounds__(256) void output_kernel(
    const __bf16* __restrict__ hf, const __bf16* __restrict__ hb,
    const __bf16* __restrict__ WoutT, const float* __restrict__ b_out,
    float* __restrict__ out)
{
  const int lane = threadIdx.x & 31;
  const int wave = threadIdx.x >> 5;          // 0..7
  const int mtile = blockIdx.x * 4 + (wave >> 1);
  const int ntile = wave & 1;
  const int marow = mtile * 16 + (lane & 15);
  const __bf16* af = hf + (size_t)marow * 256;
  const __bf16* ab = hb + (size_t)marow * 256;
  const __bf16* brow = WoutT + (size_t)(ntile * 16 + (lane & 15)) * 512;
  v8f acc = {};
#pragma unroll
  for (int kk = 0; kk < 16; ++kk) {
    const __bf16* ar = (kk < 8) ? af : ab;    // first 256 K = fwd, next 256 = bwd
    acc = wmma_bf16(load_a_frag(ar, kk & 7, lane), load_b_frag(brow, kk, lane), acc);
  }
  const int ncol = ntile * 16 + (lane & 15);
  const int mh = (lane >> 4) << 3;
  if (ncol < 20) {
    const float bias = b_out[ncol];
#pragma unroll
    for (int j = 0; j < 8; ++j)
      out[(size_t)(mtile * 16 + mh + j) * 20 + ncol] = acc[j] + bias;
  }
}

// ---------------- host-side orchestration ----------------
extern "C" void kernel_launch(void* const* d_in, const int* in_sizes, int n_in,
                              void* d_out, int out_size, void* d_ws, size_t ws_size,
                              hipStream_t stream)
{
  (void)in_sizes; (void)n_in; (void)out_size; (void)ws_size;
  const int* char_ids  = (const int*)d_in[0];
  const int* word_ids  = (const int*)d_in[1];
  const int* word_lens = (const int*)d_in[2];
  const int* lengths   = (const int*)d_in[3];
  const float* char_emb = (const float*)d_in[4];
  const float* word_emb = (const float*)d_in[5];
  const float* Wc[2]   = {(const float*)d_in[6],  (const float*)d_in[14]};
  const float* bcv[2]  = {(const float*)d_in[7],  (const float*)d_in[15]};
  const float* Wi[2]   = {(const float*)d_in[8],  (const float*)d_in[16]};
  const float* biv[2]  = {(const float*)d_in[9],  (const float*)d_in[17]};
  const float* Wcw[2]  = {(const float*)d_in[10], (const float*)d_in[18]};
  const float* bcwv[2] = {(const float*)d_in[11], (const float*)d_in[19]};
  const float* Wg[2]   = {(const float*)d_in[12], (const float*)d_in[20]};
  const float* bgv[2]  = {(const float*)d_in[13], (const float*)d_in[21]};
  const float* W_out   = (const float*)d_in[22];
  const float* b_out   = (const float*)d_in[23];

  char* ws = (char*)d_ws;
  float* out = (float*)d_out;

  __bf16* chz   = (__bf16*)(ws + OFF_CHZ);
  __bf16* wez   = (__bf16*)(ws + OFF_WEZ);
  float*  acoef = (float*)(ws + OFF_ACOEF);
  float*  hasb  = (float*)(ws + OFF_HAS);
  __bf16* WoutT = (__bf16*)(ws + OFF_WOUTT);

  prep_kernel<<<16384, 128, 0, stream>>>(char_ids, word_ids, word_lens,
                                         char_emb, word_emb, chz, wez, acoef, hasb);

  for (int d = 0; d < 2; ++d) {
    char* db = ws + OFF_DIR + (size_t)d * DIR_STRIDE;
    __bf16* WxhT  = (__bf16*)(db + D_WXHT);
    __bf16* WiT   = (__bf16*)(db + D_WIT);
    __bf16* WcwT  = (__bf16*)(db + D_WCWT);
    __bf16* WgT   = (__bf16*)(db + D_WGT);
    __bf16* wgsum = (__bf16*)(db + D_WGSUM);
    float*  wcsum = (float*)(db + D_WCSUM);
    float*  wgate = (float*)(db + D_WGATE);

    packWxh_kernel<<<1536, 256, 0, stream>>>(Wc[d], WxhT);
    packT_kernel<<<128, 256, 0, stream>>>(Wi[d],  WiT,  256, 256, 128, 100);
    packT_kernel<<<128, 256, 0, stream>>>(Wcw[d], WcwT, 256, 256, 128, 100);
    packT_kernel<<<256, 256, 0, stream>>>(Wg[d],  WgT,  256, 256, 256, 256);
    wordproj_kernel<<<4096, 512, 0, stream>>>(wez, acoef, WiT, WcwT,
                                              biv[d], bcwv[d], wgsum, wcsum);
    wgate_kernel<<<1024, 512, 0, stream>>>(wgsum, WgT, bgv[d], wgate);
  }
  packT_kernel<<<64, 256, 0, stream>>>(W_out, WoutT, 32, 20, 512, 512);

  recurrence_kernel<<<2, 1024, 0, stream>>>(ws, bcv[0], bcv[1], lengths);

  __bf16* hf = (__bf16*)(ws + OFF_DIR + D_HOUT);
  __bf16* hb = (__bf16*)(ws + OFF_DIR + DIR_STRIDE + D_HOUT);
  output_kernel<<<256, 256, 0, stream>>>(hf, hb, WoutT, b_out, out);
}
